// CoAttention_layer_16166256903001
// MI455X (gfx1250) — matrix operand order repaired and checked by reading, not compiled, not run on hardware
//
#include <hip/hip_runtime.h>
#include <hip/hip_bf16.h>

#define B_ 16
#define N_ 8
#define L_ 64
#define S_ 32
#define D_ 256
#define H_ 128

typedef __attribute__((ext_vector_type(16))) __bf16 v16bf;
typedef __attribute__((ext_vector_type(8)))  float  v8f;

// ---------------------------------------------------------------------------
// C[M,ND] = A[M,KD] * Bm[ND,KD]^T (+ bias[ND]); f32 in/out, bf16 WMMA
// 16x16x32 with f32 accumulate.  Block = 128 threads (4 waves).  Each wave
// owns ONE 16-row tile and FOUR independent 16-col tiles (4 accumulator
// chains -> WMMA latency hiding, A operand reused 4x).  K fully unrolled.
// A and B use contiguous per-lane k order k = 16*half + s (wave32).
// ---------------------------------------------------------------------------
template <int KD, int ND, bool HAS_BIAS>
__global__ __launch_bounds__(128) void gemm_bf16_wmma(
    const float* __restrict__ A, const float* __restrict__ Bm,
    const float* __restrict__ bias, float* __restrict__ C)
{
  const int lane = threadIdx.x & 31;
  const int wave = threadIdx.x >> 5;   // 0..3
  const int half = lane >> 4;          // 0: lanes 0-15, 1: lanes 16-31
  const int l16  = lane & 15;

  const int tm  = blockIdx.x;          // 16-row tile index
  const int arow = tm * 16 + l16;      // A: lane holds row M = lane%16

  // 4 column tiles per wave; ND/16 tiles total, 4 waves * 4 tiles = ND/16
  int bcol[4];
  const float* bptr[4];
#pragma unroll
  for (int j = 0; j < 4; ++j) {
    bcol[j] = (wave * 4 + j) * 16 + l16;          // B: lane holds col N
    bptr[j] = Bm + (size_t)bcol[j] * KD;
  }
  const float* aptr = A + (size_t)arow * KD;

  v8f acc0 = {}, acc1 = {}, acc2 = {}, acc3 = {};

#pragma unroll
  for (int k0 = 0; k0 < KD; k0 += 32) {
    // A 16x32 bf16: K = k0 + half*16 + t
    const float* ap0 = aptr + k0 + half * 16;
    const float* ap1 = ap0 + 8;
    v16bf a;
#pragma unroll
    for (int t = 0; t < 8; ++t) {
      a[t]     = (__bf16)ap0[t];
      a[t + 8] = (__bf16)ap1[t];
    }
    // B 32x16 bf16: element (k,n) = Bm[n*KD + k]; lane k = k0 + half*16 + t
    v16bf b0, b1, b2, b3;
    const float* bp0 = bptr[0] + k0 + half * 16;
    const float* bp1 = bptr[1] + k0 + half * 16;
    const float* bp2 = bptr[2] + k0 + half * 16;
    const float* bp3 = bptr[3] + k0 + half * 16;
#pragma unroll
    for (int t = 0; t < 16; ++t) {
      b0[t] = (__bf16)bp0[t];
      b1[t] = (__bf16)bp1[t];
      b2[t] = (__bf16)bp2[t];
      b3[t] = (__bf16)bp3[t];
    }
    acc0 = __builtin_amdgcn_wmma_f32_16x16x32_bf16(false, a, false, b0,
                                                   (short)0, acc0, false, false);
    acc1 = __builtin_amdgcn_wmma_f32_16x16x32_bf16(false, a, false, b1,
                                                   (short)0, acc1, false, false);
    acc2 = __builtin_amdgcn_wmma_f32_16x16x32_bf16(false, a, false, b2,
                                                   (short)0, acc2, false, false);
    acc3 = __builtin_amdgcn_wmma_f32_16x16x32_bf16(false, a, false, b3,
                                                   (short)0, acc3, false, false);
  }

  // C/D layout: VGPR r -> row = half*8 + r, col = lane%16.
  // Base pointer computed once; stores use constant row offsets.
  float bb0 = 0.f, bb1 = 0.f, bb2 = 0.f, bb3 = 0.f;
  if (HAS_BIAS) {
    bb0 = bias[bcol[0]]; bb1 = bias[bcol[1]];
    bb2 = bias[bcol[2]]; bb3 = bias[bcol[3]];
  }
  float* crow = C + (size_t)(tm * 16 + half * 8) * ND;
#pragma unroll
  for (int r = 0; r < 8; ++r) {
    crow[(size_t)r * ND + bcol[0]] = acc0[r] + bb0;
    crow[(size_t)r * ND + bcol[1]] = acc1[r] + bb1;
    crow[(size_t)r * ND + bcol[2]] = acc2[r] + bb2;
    crow[(size_t)r * ND + bcol[3]] = acc3[r] + bb3;
  }
}

// ---------------------------------------------------------------------------
// Fused co-attention per (b,n): scores -> softmax over L -> mean_s folded into
// abar[l] -> sr[b,n,d] = sum_l abar[l] * x[b,n,l,d].
// Block = 256 threads, one block per (b,n).  Wx tile staged in LDS (stride 257
// padding: column-strided reads hit 32 distinct banks).
// ---------------------------------------------------------------------------
#define WXS 257
__global__ __launch_bounds__(256) void coatt_kernel(
    const float* __restrict__ Wx, const float* __restrict__ Us,
    const float* __restrict__ v,  const float* __restrict__ x,
    float* __restrict__ sr)
{
  extern __shared__ float sm[];
  float* wx_sh = sm;                    // 64 * 257
  float* v_sh  = wx_sh + L_ * WXS;      // 256
  float* us_sh = v_sh + D_;             // 256
  float* part  = us_sh + D_;            // 256
  float* red   = part + 256;            // 64
  float* abar  = red + 64;              // 64

  const int tid = threadIdx.x;
  const int bn  = blockIdx.x;           // b*N + n
  const int b   = bn >> 3;

  const float* wxg = Wx + (size_t)bn * (L_ * D_);
  for (int i = tid; i < L_ * D_; i += 256) {
    int l = i >> 8, d = i & (D_ - 1);
    wx_sh[l * WXS + d] = wxg[i];
  }
  v_sh[tid] = v[tid];
  if (tid < L_) abar[tid] = 0.f;
  __syncthreads();

  const int g = tid >> 6;               // 0..3 : 64-wide d-slice
  const int l = tid & 63;

  for (int s = 0; s < S_; ++s) {
    us_sh[tid] = Us[((size_t)(b * S_ + s)) * D_ + tid];
    __syncthreads();
    float p = 0.f;
    const float* wrow = wx_sh + l * WXS + g * 64;
    const float* urow = us_sh + g * 64;
    const float* vrow = v_sh + g * 64;
#pragma unroll 4
    for (int j = 0; j < 64; ++j)
      p += vrow[j] * tanhf(wrow[j] + urow[j]);
    part[tid] = p;
    __syncthreads();
    if (tid < 64)
      red[tid] = part[tid] + part[64 + tid] + part[128 + tid] + part[192 + tid];
    __syncthreads();
    float m = -1e30f;
    for (int i2 = 0; i2 < 64; ++i2) m = fmaxf(m, red[i2]);
    if (tid < 64) part[tid] = expf(red[tid] - m);
    __syncthreads();
    float sum = 0.f;
    for (int i2 = 0; i2 < 64; ++i2) sum += part[i2];
    if (tid < 64) abar[tid] += part[tid] / (sum * (float)S_);
    __syncthreads();
  }

  // sr[b,n,d] = sum_l abar[l] * x[b,n,l,d]   (coalesced over d = tid)
  float acc = 0.f;
  const float* xb = x + (size_t)bn * (L_ * D_);
#pragma unroll 4
  for (int ll = 0; ll < L_; ++ll) acc += abar[ll] * xb[ll * D_ + tid];
  sr[(size_t)bn * D_ + tid] = acc;
}

// ---------------------------------------------------------------------------
// BiLSTM over the N axis: 2 blocks (dir 0 = fwd, dir 1 = bwd), 1024 threads.
// Gate order i,f,g,o.  h/c/gates/u kept in LDS; weights stream from L2.
// ---------------------------------------------------------------------------
__global__ __launch_bounds__(1024) void lstm_kernel(
    const float* __restrict__ sr,
    const float* __restrict__ Wih_f, const float* __restrict__ Whh_f,
    const float* __restrict__ bih_f, const float* __restrict__ bhh_f,
    const float* __restrict__ Wih_b, const float* __restrict__ Whh_b,
    const float* __restrict__ bih_b, const float* __restrict__ bhh_b,
    float* __restrict__ hseq)
{
  extern __shared__ float sm[];
  float* u_sh = sm;                     // 16 * 256
  float* h_sh = u_sh + B_ * D_;         // 16 * 128
  float* c_sh = h_sh + B_ * H_;         // 16 * 128
  float* g_sh = c_sh + B_ * H_;         // 16 * 512

  const int tid = threadIdx.x;
  const int dir = blockIdx.x;
  const float* Wih = dir ? Wih_b : Wih_f;
  const float* Whh = dir ? Whh_b : Whh_f;
  const float* bih = dir ? bih_b : bih_f;
  const float* bhh = dir ? bhh_b : bhh_f;

  for (int i = tid; i < B_ * H_; i += 1024) { h_sh[i] = 0.f; c_sh[i] = 0.f; }
  __syncthreads();

  for (int step = 0; step < N_; ++step) {
    const int n = dir ? (N_ - 1 - step) : step;
    for (int i = tid; i < B_ * D_; i += 1024) {
      int b = i >> 8, d = i & (D_ - 1);
      u_sh[i] = sr[((size_t)b * N_ + n) * D_ + d];
    }
    __syncthreads();
    // gates[b][j] = u·Wih[j] + h·Whh[j] + bih[j] + bhh[j]
    for (int p = tid; p < B_ * 4 * H_; p += 1024) {
      int b = p >> 9, j = p & 511;
      float acc = bih[j] + bhh[j];
      const float* wr = Wih + (size_t)j * D_;
      const float* ur = u_sh + b * D_;
#pragma unroll 4
      for (int d = 0; d < D_; ++d) acc += ur[d] * wr[d];
      const float* vr = Whh + (size_t)j * H_;
      const float* hr = h_sh + b * H_;
#pragma unroll 4
      for (int k = 0; k < H_; ++k) acc += hr[k] * vr[k];
      g_sh[b * 512 + j] = acc;
    }
    __syncthreads();
    for (int p = tid; p < B_ * H_; p += 1024) {
      int b = p >> 7, k = p & (H_ - 1);
      float gi = g_sh[b * 512 + k];
      float gf = g_sh[b * 512 + 128 + k];
      float gg = g_sh[b * 512 + 256 + k];
      float go = g_sh[b * 512 + 384 + k];
      float si = 1.f / (1.f + expf(-gi));
      float sf = 1.f / (1.f + expf(-gf));
      float so = 1.f / (1.f + expf(-go));
      float cn = sf * c_sh[p] + si * tanhf(gg);
      float hn = so * tanhf(cn);
      c_sh[p] = cn;
      h_sh[p] = hn;
      hseq[(((size_t)dir * N_ + n) * B_ + b) * H_ + k] = hn;
    }
    __syncthreads();
  }
}

// ---------------------------------------------------------------------------
// c[b,n,:] = concat(hf[n,b,:], hb[n,b,:])  -> cmat[(b*N+n), 256]
// ---------------------------------------------------------------------------
__global__ void build_c_kernel(const float* __restrict__ hseq,
                               float* __restrict__ cmat)
{
  int i = blockIdx.x * blockDim.x + threadIdx.x;
  if (i >= B_ * N_ * D_) return;
  int bn = i >> 8, d = i & (D_ - 1);
  int b = bn >> 3, n = bn & 7;
  float vv;
  if (d < H_) vv = hseq[((size_t)n * B_ + b) * H_ + d];
  else        vv = hseq[(((size_t)N_ + n) * B_ + b) * H_ + (d - H_)];
  cmat[i] = vv;
}

// ---------------------------------------------------------------------------
// Self-attention pooling: att[b,n] = tanh(attpre[b,n,:])·sav; softmax over n;
// out[b,:] = sum_n att * c[b,n,:].  One block per b; wave w handles n = w.
// ---------------------------------------------------------------------------
__global__ __launch_bounds__(256) void pool_kernel(
    const float* __restrict__ attpre, const float* __restrict__ sav,
    const float* __restrict__ cmat, float* __restrict__ out)
{
  __shared__ float att_sh[N_];
  const int tid = threadIdx.x;
  const int b = blockIdx.x;
  const int w = tid >> 5, lane = tid & 31;

  float p = 0.f;
  const float* row = attpre + (size_t)(b * N_ + w) * D_;
  for (int d = lane; d < D_; d += 32) p += tanhf(row[d]) * sav[d];
#pragma unroll
  for (int off = 16; off; off >>= 1) p += __shfl_xor(p, off, 32);
  if (lane == 0) att_sh[w] = p;
  __syncthreads();

  float m = -1e30f;
  for (int n = 0; n < N_; ++n) m = fmaxf(m, att_sh[n]);
  float e[N_], sum = 0.f;
  for (int n = 0; n < N_; ++n) { e[n] = expf(att_sh[n] - m); sum += e[n]; }
  float acc = 0.f;
  for (int n = 0; n < N_; ++n)
    acc += (e[n] / sum) * cmat[(size_t)(b * N_ + n) * D_ + tid];
  out[(size_t)b * D_ + tid] = acc;
}

// ---------------------------------------------------------------------------
extern "C" void kernel_launch(void* const* d_in, const int* in_sizes, int n_in,
                              void* d_out, int out_size, void* d_ws, size_t ws_size,
                              hipStream_t stream)
{
  const float* x     = (const float*)d_in[0];
  const float* s     = (const float*)d_in[1];
  const float* W     = (const float*)d_in[2];
  const float* U     = (const float*)d_in[3];
  const float* v     = (const float*)d_in[4];
  const float* Wih_f = (const float*)d_in[5];
  const float* Whh_f = (const float*)d_in[6];
  const float* bih_f = (const float*)d_in[7];
  const float* bhh_f = (const float*)d_in[8];
  const float* Wih_b = (const float*)d_in[9];
  const float* Whh_b = (const float*)d_in[10];
  const float* bih_b = (const float*)d_in[11];
  const float* bhh_b = (const float*)d_in[12];
  const float* saW   = (const float*)d_in[13];
  const float* sab   = (const float*)d_in[14];
  const float* sav   = (const float*)d_in[15];
  float* out = (float*)d_out;

  float* ws   = (float*)d_ws;
  float* Wx   = ws;                       // 8192*256  = 2097152
  float* Usb  = Wx + 2097152;             // 512*256   = 131072
  float* sr   = Usb + 131072;             // 128*256   = 32768
  float* hseq = sr + 32768;               // 2*8*16*128= 32768
  float* cmat = hseq + 32768;             // 128*256   = 32768
  float* attp = cmat + 32768;             // 128*256   = 32768

  // Wx = x @ W^T   [8192,256] x [256,256]^T
  gemm_bf16_wmma<256, 256, false><<<512, 128, 0, stream>>>(x, W, nullptr, Wx);
  // Us = s @ U^T   [512,256] x [256,256]^T
  gemm_bf16_wmma<256, 256, false><<<32, 128, 0, stream>>>(s, U, nullptr, Usb);

  // fused co-attention -> sr[b,n,d]
  size_t co_lds = (size_t)(L_ * WXS + D_ + D_ + 256 + 64 + 64) * sizeof(float);
  coatt_kernel<<<B_ * N_, 256, co_lds, stream>>>(Wx, Usb, v, x, sr);

  // BiLSTM over N
  size_t lstm_lds = (size_t)(B_ * D_ + 2 * B_ * H_ + B_ * 4 * H_) * sizeof(float);
  lstm_kernel<<<2, 1024, lstm_lds, stream>>>(sr, Wih_f, Whh_f, bih_f, bhh_f,
                                             Wih_b, Whh_b, bih_b, bhh_b, hseq);

  // c = concat(hf, hb)
  build_c_kernel<<<(B_ * N_ * D_ + 255) / 256, 256, 0, stream>>>(hseq, cmat);

  // attpre = c @ saW^T + sab
  gemm_bf16_wmma<256, 256, true><<<8, 128, 0, stream>>>(cmat, saW, sab, attp);

  // pooled output g[b, 2H]
  pool_kernel<<<B_, 256, 0, stream>>>(attp, sav, cmat, out);
}
